// MultiHeadAttention_60799557042884
// MI455X (gfx1250) — matrix operand order repaired and checked by reading, not compiled, not run on hardware
//
#include <hip/hip_runtime.h>

typedef _Float16 f16;
typedef __attribute__((ext_vector_type(16))) _Float16 v16h;
typedef __attribute__((ext_vector_type(8)))  float    v8f;

// Problem constants (bs=16, L=1024, dq=dk=dv=128, da=512, do=128, H=8, DH=64)
#define BS   16
#define LSEQ 1024
#define DIN  128
#define DA   512
#define NH   8
#define DHD  64
#define DOUT 128

#define QSCALE 0.125f      // 1/sqrt(64)
#define NEGV  (-1e9f)

// ---- CDNA5 dependency-counter waits (SOPP, ISA ch.15.5) ----
#define WAIT_ASYNC0 asm volatile("s_wait_asynccnt 0x0" ::: "memory")
#define WAIT_DS0    asm volatile("s_wait_dscnt 0x0" ::: "memory")

// ---- async global->LDS b128 (ASYNCcnt-tracked, CDNA5 ISA) ----
// VGLOBAL encoding: VDST = wave-relative LDS byte offset in a VGPR,
// address = 64-bit VGPR pair, SADDR = off.  Inline asm avoids the toolchain's
// builtin prototype (which wants v4i* params of uncertain order).
__device__ __forceinline__ void async_b128(const f16* g, f16* l) {
  // Generic LDS pointer: low 32 bits are the wave-relative LDS byte offset.
  unsigned loff = (unsigned)(unsigned long long)(void*)l;
  asm volatile("global_load_async_to_lds_b128 %0, %1, off"
               :: "v"(loff), "v"(g) : "memory");
}

__device__ __forceinline__ v8f wmma16(v16h a, v16h b, v8f c) {
  // D = A(16x32 f16) * B(32x16 f16) + C(16x16 f32)
  return __builtin_amdgcn_wmma_f32_16x16x32_f16(
      /*neg_a=*/false, a, /*neg_b=*/false, b,
      /*c_mod=*/(short)0, c, /*reuse_a=*/false, /*reuse_b=*/false);
}

// ---------------------------------------------------------------------------
// Kernel 1: f32 (b, DIN, L) -> f16 (b, L, DIN) transpose+convert.
// ---------------------------------------------------------------------------
__global__ void cvt_transpose(const float* __restrict__ x, f16* __restrict__ xT) {
  size_t idx = (size_t)blockIdx.x * blockDim.x + threadIdx.x;
  int l = (int)(idx % LSEQ);
  size_t t = idx / LSEQ;
  int c = (int)(t % DIN);
  int b = (int)(t / DIN);
  xT[((size_t)b * LSEQ + l) * DIN + c] = (f16)x[idx];
}

// ---------------------------------------------------------------------------
// Kernel 2: projection GEMM  Y[o,l] = sum_c W[o,c] * x[b,c,l] + bias[o]
// mode 0: Q -> (b,h,L,64) f16, pre-scaled by 1/sqrt(DH)
// mode 1: K -> (b,h,L,64) f16
// mode 2: V -> (b,512,L)  f16   (key-fastest for the P@V B-operand)
// ---------------------------------------------------------------------------
__global__ void proj_wmma(const f16* __restrict__ xT, const float* __restrict__ W,
                          const float* __restrict__ bias, f16* __restrict__ out, int mode) {
  const int lane = threadIdx.x;
  const int half = lane >> 4, lr = lane & 15;
  const int obase = blockIdx.x * 16, lbase = blockIdx.y * 16, b = blockIdx.z;

  v8f acc;
  #pragma unroll
  for (int e = 0; e < 8; ++e) acc[e] = 0.0f;

  const float* wrow = W + (size_t)(obase + lr) * DIN;
  const f16*   xrow = xT + ((size_t)b * LSEQ + lbase + lr) * DIN + 16 * half;

  #pragma unroll
  for (int kc = 0; kc < DIN; kc += 32) {
    v16h a;
    #pragma unroll
    for (int e = 0; e < 16; ++e) {
      int k = kc + 16 * half + e;                        // A: k = e + 16*half
      a[e] = (f16)wrow[k];
    }
    v16h bm = *(const v16h*)(xrow + kc);                 // B: k = e + 16*half
    acc = wmma16(a, bm, acc);
  }

  const int l = lbase + lr;
  #pragma unroll
  for (int v = 0; v < 8; ++v) {
    int o = obase + v + 8 * half;                        // C/D: row = v + 8*half
    float y = acc[v] + bias[o];
    if (mode == 0) y *= QSCALE;
    if (mode == 2) {
      out[((size_t)b * DA + o) * LSEQ + l] = (f16)y;
    } else {
      int h = o >> 6, d = o & 63;
      out[(((size_t)b * NH + h) * LSEQ + l) * DHD + d] = (f16)y;
    }
  }
}

// ---------------------------------------------------------------------------
// Kernel 3: flash attention with async double-buffered K/V staging in LDS.
// One wave per (b, h, 16-query tile). Per 32-key chunk:
//   wait asynccnt -> LDS reads of K/V B-operands -> wait dscnt ->
//   issue async prefetch of next chunk -> 4 WMMAs (S) -> online softmax ->
//   P relayout via LDS -> 4 WMMAs (O += P*V).
// ---------------------------------------------------------------------------
__device__ __forceinline__ void stage_kv(const f16* kbasep, const f16* vbasep,
                                         int kb, f16* sKb, f16* sVb, int lane) {
  // K tile: 32 rows x 64 f16 (row-major) = 256 x 16B segments, coalesced.
  #pragma unroll
  for (int i = 0; i < 8; ++i) {
    int idx = i * 32 + lane;
    async_b128(kbasep + (size_t)(kb + (idx >> 3)) * DHD + (idx & 7) * 8,
               sKb + idx * 8);
  }
  // V tile: 64 dims x 32 keys (dim-major), 4 x 16B segments per dim.
  #pragma unroll
  for (int i = 0; i < 8; ++i) {
    int idx = i * 32 + lane;
    async_b128(vbasep + (size_t)(idx >> 2) * LSEQ + kb + (idx & 3) * 8,
               sVb + idx * 8);
  }
}

__global__ void attn_wmma(const f16* __restrict__ qh, const f16* __restrict__ kh,
                          const f16* __restrict__ vh, const unsigned char* __restrict__ mask,
                          f16* __restrict__ attn) {
  __shared__ __align__(64) f16  sK[2][32 * 64];   // [buf][row*64 + d]      4KB each
  __shared__ __align__(64) f16  sV[2][64 * 32];   // [buf][dim*32 + key]    4KB each
  __shared__ __align__(64) float Pld[16 * 32];

  const int lane = threadIdx.x, half = lane >> 4, lr = lane & 15;
  const int qbase = blockIdx.x * 16, h = blockIdx.y, b = blockIdx.z;

  // Q A-operands (16x64 split into two 16x32), already scaled by 1/sqrt(DH)
  const f16* qrow = qh + (((size_t)b * NH + h) * LSEQ + qbase + lr) * DHD;
  v16h aQ0, aQ1;
  #pragma unroll
  for (int e = 0; e < 16; ++e) {
    int k = 16 * half + e;
    aQ0[e] = qrow[k];
    aQ1[e] = qrow[32 + k];
  }

  unsigned rm = 0;
  #pragma unroll
  for (int v = 0; v < 8; ++v)
    rm |= mask[(size_t)b * LSEQ + qbase + 8 * half + v] ? (1u << v) : 0u;

  float mrow[8], lrow[8];
  v8f oacc[4];
  #pragma unroll
  for (int v = 0; v < 8; ++v) { mrow[v] = -1e30f; lrow[v] = 0.0f; }
  #pragma unroll
  for (int d = 0; d < 4; ++d)
    #pragma unroll
    for (int e = 0; e < 8; ++e) oacc[d][e] = 0.0f;

  const f16* kbasep = kh + (((size_t)b * NH + h) * LSEQ) * DHD;
  const f16* vbasep = vh + ((size_t)b * DA + h * DHD) * LSEQ;

  // Prologue: stage chunk 0.
  stage_kv(kbasep, vbasep, 0, sK[0], sV[0], lane);

  for (int kb = 0; kb < LSEQ; kb += 32) {
    const int buf = (kb >> 5) & 1;
    WAIT_ASYNC0;                              // chunk kb resident in LDS

    // Pull this chunk's B-operands into registers (DS reads).
    const f16* kcur = sK[buf];
    const f16* vcur = sV[buf];
    v16h bK[4];                               // [t*2 + c]
    #pragma unroll
    for (int t = 0; t < 2; ++t)
      #pragma unroll
      for (int c = 0; c < 2; ++c)
        bK[t * 2 + c] = *(const v16h*)(kcur + (t * 16 + lr) * DHD + c * 32 + 16 * half);
    v16h bV[4];
    #pragma unroll
    for (int d = 0; d < 4; ++d)
      bV[d] = *(const v16h*)(vcur + (d * 16 + lr) * 32 + 16 * half);
    WAIT_DS0;                                 // reads done before async overwrites

    // Prefetch next chunk; overlaps with all compute below.
    if (kb + 32 < LSEQ)
      stage_kv(kbasep, vbasep, kb + 32, sK[buf ^ 1], sV[buf ^ 1], lane);

    // ---- S = Q * K^T  (16 rows x 32 keys, two 16x16 D tiles) ----
    v8f s0, s1;
    #pragma unroll
    for (int e = 0; e < 8; ++e) { s0[e] = 0.0f; s1[e] = 0.0f; }
    s0 = wmma16(aQ0, bK[0], s0); s0 = wmma16(aQ1, bK[1], s0);
    s1 = wmma16(aQ0, bK[2], s1); s1 = wmma16(aQ1, bK[3], s1);

    const bool cm0 = mask[(size_t)b * LSEQ + kb + lr] != 0;
    const bool cm1 = mask[(size_t)b * LSEQ + kb + 16 + lr] != 0;

    // ---- mask + row max (a row's 16 columns live in one 16-lane half) ----
    float t[8];
    #pragma unroll
    for (int v = 0; v < 8; ++v) {
      bool rmv = (rm >> v) & 1;
      float a0 = (rmv || cm0) ? NEGV : s0[v];
      float a1 = (rmv || cm1) ? NEGV : s1[v];
      s0[v] = a0; s1[v] = a1;
      t[v] = fmaxf(a0, a1);
    }
    #pragma unroll
    for (int m = 1; m <= 8; m <<= 1)
      #pragma unroll
      for (int v = 0; v < 8; ++v)
        t[v] = fmaxf(t[v], __shfl_xor(t[v], m, 32));

    // ---- online softmax update ----
    float rs[8];
    #pragma unroll
    for (int v = 0; v < 8; ++v) {
      float mnew = fmaxf(mrow[v], t[v]);
      float sc = __expf(mrow[v] - mnew);
      mrow[v] = mnew;
      float p0 = __expf(s0[v] - mnew);
      float p1 = __expf(s1[v] - mnew);
      rs[v] = p0 + p1;
      lrow[v] *= sc;
      #pragma unroll
      for (int d = 0; d < 4; ++d) oacc[d][v] *= sc;
      Pld[(v + 8 * half) * 32 + lr]      = p0;
      Pld[(v + 8 * half) * 32 + lr + 16] = p1;
    }
    #pragma unroll
    for (int m = 1; m <= 8; m <<= 1)
      #pragma unroll
      for (int v = 0; v < 8; ++v)
        rs[v] += __shfl_xor(rs[v], m, 32);
    #pragma unroll
    for (int v = 0; v < 8; ++v) lrow[v] += rs[v];

    // ---- relayout P (C layout) -> A operand via LDS ----
    WAIT_DS0;
    v16h aP;
    #pragma unroll
    for (int e = 0; e < 16; ++e) {
      int k = 16 * half + e;
      aP[e] = (f16)Pld[lr * 32 + k];
    }

    // ---- O += P * V  (four 16x16 dim tiles) ----
    #pragma unroll
    for (int d = 0; d < 4; ++d)
      oacc[d] = wmma16(aP, bV[d], oacc[d]);
  }

  // ---- finalize: divide by row sums, zero fully-masked rows, store f16 ----
  float inv[8];
  #pragma unroll
  for (int v = 0; v < 8; ++v) {
    float li = lrow[v];
    float iv = (li > 0.0f) ? 1.0f / li : 0.0f;
    if ((rm >> v) & 1) iv = 0.0f;
    inv[v] = iv;
  }
  #pragma unroll
  for (int d = 0; d < 4; ++d)
    #pragma unroll
    for (int v = 0; v < 8; ++v) {
      int l = qbase + v + 8 * half;
      attn[((size_t)b * LSEQ + l) * DA + h * DHD + d * 16 + lr] = (f16)(oacc[d][v] * inv[v]);
    }
}

// ---------------------------------------------------------------------------
// Kernel 4: output GEMM  out[o,l] = sum_c w_o[o,c] * attn[b,l,c]  (f32 out)
// ---------------------------------------------------------------------------
__global__ void out_wmma(const f16* __restrict__ attn, const float* __restrict__ Wo,
                         float* __restrict__ out) {
  const int lane = threadIdx.x, half = lane >> 4, lr = lane & 15;
  const int obase = blockIdx.x * 16, lbase = blockIdx.y * 16, b = blockIdx.z;

  v8f acc;
  #pragma unroll
  for (int e = 0; e < 8; ++e) acc[e] = 0.0f;

  const float* wrow = Wo + (size_t)(obase + lr) * DA;
  const f16*   arow = attn + ((size_t)b * LSEQ + lbase + lr) * DA + 16 * half;

  #pragma unroll
  for (int kc = 0; kc < DA; kc += 32) {
    v16h a;
    #pragma unroll
    for (int e = 0; e < 16; ++e) {
      int k = kc + 16 * half + e;
      a[e] = (f16)wrow[k];
    }
    v16h bm = *(const v16h*)(arow + kc);
    acc = wmma16(a, bm, acc);
  }

  const int l = lbase + lr;
  #pragma unroll
  for (int v = 0; v < 8; ++v) {
    int o = obase + v + 8 * half;
    out[((size_t)b * DOUT + o) * LSEQ + l] = acc[v];
  }
}

// ---------------------------------------------------------------------------
extern "C" void kernel_launch(void* const* d_in, const int* in_sizes, int n_in,
                              void* d_out, int out_size, void* d_ws, size_t ws_size,
                              hipStream_t stream) {
  (void)in_sizes; (void)n_in; (void)out_size; (void)ws_size;

  const float* q   = (const float*)d_in[0];
  const float* k   = (const float*)d_in[1];
  const float* v   = (const float*)d_in[2];
  const unsigned char* mask = (const unsigned char*)d_in[3];
  const float* w_q = (const float*)d_in[4];
  const float* b_q = (const float*)d_in[5];
  const float* w_k = (const float*)d_in[6];
  const float* b_k = (const float*)d_in[7];
  const float* w_v = (const float*)d_in[8];
  const float* b_v = (const float*)d_in[9];
  const float* w_o = (const float*)d_in[10];
  float* out = (float*)d_out;

  char* ws = (char*)d_ws;
  size_t off = 0;
  auto carve = [&](size_t bytes) -> void* {
    void* p = ws + off;
    off += (bytes + 255) & ~(size_t)255;
    return p;
  };
  f16* qT    = (f16*)carve((size_t)BS * LSEQ * DIN * sizeof(f16));
  f16* kT    = (f16*)carve((size_t)BS * LSEQ * DIN * sizeof(f16));
  f16* vT    = (f16*)carve((size_t)BS * LSEQ * DIN * sizeof(f16));
  f16* qhp   = (f16*)carve((size_t)BS * DA * LSEQ * sizeof(f16));
  f16* khp   = (f16*)carve((size_t)BS * DA * LSEQ * sizeof(f16));
  f16* vhp   = (f16*)carve((size_t)BS * DA * LSEQ * sizeof(f16));
  f16* attnp = (f16*)carve((size_t)BS * LSEQ * DA * sizeof(f16));

  const int nT = BS * DIN * LSEQ;
  cvt_transpose<<<nT / 256, 256, 0, stream>>>(q, qT);
  cvt_transpose<<<nT / 256, 256, 0, stream>>>(k, kT);
  cvt_transpose<<<nT / 256, 256, 0, stream>>>(v, vT);

  dim3 gp(DA / 16, LSEQ / 16, BS);
  proj_wmma<<<gp, 32, 0, stream>>>(qT, w_q, b_q, qhp, 0);
  proj_wmma<<<gp, 32, 0, stream>>>(kT, w_k, b_k, khp, 1);
  proj_wmma<<<gp, 32, 0, stream>>>(vT, w_v, b_v, vhp, 2);

  dim3 ga(LSEQ / 16, NH, BS);
  attn_wmma<<<ga, 32, 0, stream>>>(qhp, khp, vhp, mask, attnp);

  dim3 go(DOUT / 16, LSEQ / 16, BS);
  out_wmma<<<go, 32, 0, stream>>>(attnp, w_o, out);
}
